// RingNet_24515673326158
// MI455X (gfx1250) — matrix operand order repaired and checked by reading, not compiled, not run on hardware
//
#include <hip/hip_runtime.h>
#include <stdint.h>

// ---------------------------------------------------------------------------
// RingNet on gfx1250: SAT ring features (LDS) -> BN1 -> GEMM1 (bf16 WMMA,
// 1248->512, +bias, ReLU) -> folded BN2 -> GEMM2 (bf16 WMMA, 512->96, +bias)
// Fragment-major LDS B tiles (all-b128 LDS traffic), double-buffered K loop
// using CDNA5 global_load_async_to_lds (ASYNCcnt), cvt_pk_bf16_f32 packing.
// ---------------------------------------------------------------------------

#define TRY_ASYNC  1
#define TRY_CVT_PK 1

typedef __attribute__((ext_vector_type(16))) __bf16 v16bf;
typedef __attribute__((ext_vector_type(8)))  float  v8f;

#define NPIX_TOT 65536           // 4 * 128 * 128
#define KDIM     1248
#define QDIM     (KDIM / 2)      // 624 k-pair rows of features
#define HDIM     512
#define HQ       (HDIM / 2)      // 256 k-pair rows of h
#define ODIM     96
#define IMG      16384           // 128*128
#define SSTRIDE  129             // padded SAT row stride (floats)
#define IMGSZ    (128 * SSTRIDE)
#define BCSTRIDE 20              // LDS B-tile column stride in dwords (16 + pad)

#if TRY_ASYNC && __has_builtin(__builtin_amdgcn_global_load_async_to_lds_b128) && \
                 __has_builtin(__builtin_amdgcn_global_load_async_to_lds_b32)
#define USE_ASYNC 1
#else
#define USE_ASYNC 0
#endif

#if USE_ASYNC
// builtins take typed address-space pointers: b128 -> int4*, b32 -> int*.
// generic->AS(1)/AS(3) via integer casts (flat LDS addr low 32 bits == offset).
typedef int async_v4i __attribute__((__vector_size__(16)));
#define AS1_B128(p) ((__attribute__((address_space(1))) async_v4i*)(unsigned long long)(p))
#define AS3_B128(p) ((__attribute__((address_space(3))) async_v4i*)(unsigned int)(unsigned long long)(p))
#define AS1_B32(p)  ((__attribute__((address_space(1))) int*)(unsigned long long)(p))
#define AS3_B32(p)  ((__attribute__((address_space(3))) int*)(unsigned int)(unsigned long long)(p))
#endif

__device__ __forceinline__ void async_wait0() {
#if USE_ASYNC
#if __has_builtin(__builtin_amdgcn_s_wait_asynccnt)
  __builtin_amdgcn_s_wait_asynccnt(0);
#else
  asm volatile("s_wait_asynccnt 0" ::: "memory");
#endif
#endif
}

// round-to-nearest-even f32 -> bf16 bits
__device__ __forceinline__ unsigned short f2bf(float f) {
  uint32_t u = __float_as_uint(f);
  uint32_t r = u + 0x7FFFu + ((u >> 16) & 1u);
  return (unsigned short)(r >> 16);
}

__device__ __forceinline__ uint32_t packbf2(float lo, float hi) {
#if TRY_CVT_PK && __has_builtin(__builtin_amdgcn_cvt_pk_bf16_f32)
  auto r = __builtin_amdgcn_cvt_pk_bf16_f32(lo, hi);
  uint32_t u;
  __builtin_memcpy(&u, &r, 4);
  return u;
#else
  return (uint32_t)f2bf(lo) | ((uint32_t)f2bf(hi) << 16);
#endif
}

// clamped box sum from an inclusive SAT with zero padding semantics
__device__ __forceinline__ float boxsum(const float* S, int y, int x, int p) {
  int r2 = min(y + p, 127), c2 = min(x + p, 127);
  int r1 = y - p - 1,       c1 = x - p - 1;
  float s22 = S[r2 * SSTRIDE + c2];
  float s12 = (r1 >= 0) ? S[r1 * SSTRIDE + c2] : 0.f;
  float s21 = (c1 >= 0) ? S[r2 * SSTRIDE + c1] : 0.f;
  float s11 = (r1 >= 0 && c1 >= 0) ? S[r1 * SSTRIDE + c1] : 0.f;
  return s22 - s12 - s21 + s11;
}

// --------------------------- prep kernels ----------------------------------
__global__ void prep_w1(const float* __restrict__ w1, unsigned short* __restrict__ W1b) {
  int i = blockIdx.x * 256 + threadIdx.x;
  if (i < HDIM * KDIM) W1b[i] = f2bf(w1[i]);
}

__global__ void prep_w2(const float* __restrict__ w2, const float* __restrict__ g2,
                        const float* __restrict__ v2, unsigned short* __restrict__ W2p) {
  int i = blockIdx.x * 256 + threadIdx.x;
  if (i < ODIM * HDIM) {
    int k = i & (HDIM - 1);
    float s2 = g2[k] * rsqrtf(v2[k] + 1e-5f);
    W2p[i] = f2bf(w2[i] * s2);
  }
}

__global__ void prep_b2(const float* __restrict__ w2, const float* __restrict__ g2,
                        const float* __restrict__ b2, const float* __restrict__ m2,
                        const float* __restrict__ v2, const float* __restrict__ cb2,
                        float* __restrict__ b2p) {
  int p = threadIdx.x;
  if (p < ODIM) {
    float acc = cb2[p];
    for (int k = 0; k < HDIM; ++k) {
      float s2 = g2[k] * rsqrtf(v2[k] + 1e-5f);
      acc += w2[p * HDIM + k] * (b2[k] - m2[k] * s2);
    }
    b2p[p] = acc;
  }
}

// --------------------- features: SAT rings + BN1 -> bf16 pairs -------------
// grid (48, 4): blockIdx.x = channel pair, blockIdx.y = batch image. 256 thr.
__global__ void features_kernel(const float* __restrict__ x,
                                const float* __restrict__ g1, const float* __restrict__ b1,
                                const float* __restrict__ m1, const float* __restrict__ v1,
                                uint32_t* __restrict__ featP) {
  extern __shared__ float sm[];
  const int tid = threadIdx.x;
  const int cpair = blockIdx.x;       // 0..47
  const int bimg  = blockIdx.y;       // 0..3
  const int c0 = cpair * 2;

  for (int i = 0; i < 128; ++i) {
    int idx = tid + i * 256;          // 0..32767
    int img = idx >> 14, off = idx & (IMG - 1);
    int row = off >> 7, col = off & 127;
    sm[img * IMGSZ + row * SSTRIDE + col] =
        x[(size_t)(bimg * 96 + c0 + img) * IMG + off];
  }
  __syncthreads();
  { // row prefix sums
    int img = tid >> 7, row = tid & 127;
    float* R = sm + img * IMGSZ + row * SSTRIDE;
    float s = 0.f;
    for (int i = 0; i < 128; ++i) { s += R[i]; R[i] = s; }
  }
  __syncthreads();
  { // column prefix sums
    int img = tid >> 7, col = tid & 127;
    float* C = sm + img * IMGSZ + col;
    float s = 0.f;
    for (int i = 0; i < 128; ++i) { s += C[i * SSTRIDE]; C[i * SSTRIDE] = s; }
  }
  __syncthreads();

  float sc[13][2], sh[13][2];
  for (int g = 0; g < 13; ++g)
    for (int e = 0; e < 2; ++e) {
      int ch = g * 96 + c0 + e;
      float s = g1[ch] * rsqrtf(v1[ch] + 1e-5f);
      sc[g][e] = s;
      sh[g][e] = b1[ch] - m1[ch] * s;
    }

  const float* S0 = sm;
  const float* S1 = sm + IMGSZ;
  for (int it = 0; it < 64; ++it) {
    int pix = it * 256 + tid;
    int y = pix >> 7, xx = pix & 127;
    uint32_t n = (uint32_t)(bimg * IMG + pix);
    float prev0 = 0.f, prev1 = 0.f;
    for (int g = 0; g < 13; ++g) {
      float f0 = boxsum(S0, y, xx, g);
      float f1 = boxsum(S1, y, xx, g);
      float r0 = (g == 0) ? f0 : f0 - prev0;
      float r1 = (g == 0) ? f1 : f1 - prev1;
      prev0 = f0; prev1 = f1;
      float o0 = r0 * sc[g][0] + sh[g][0];
      float o1 = r1 * sc[g][1] + sh[g][1];
      featP[(size_t)(g * 48 + cpair) * NPIX_TOT + n] = packbf2(o0, o1);
    }
  }
}

// --------------------------- GEMM1: 512 x 65536 x 1248 ---------------------
// Block tile 128(M) x 128(N); 8 waves as 4(M) x 2(N); wave tile 32x64 -> 8 WMMA.
// LDS A: [row][k] stride 40 bf16. LDS B: fragment-major [col][kpair] stride 20 dw.
__global__ __launch_bounds__(256) void gemm1_kernel(
    const unsigned short* __restrict__ W1b,   // [512][1248] bf16 row-major
    const uint32_t* __restrict__ featP,       // [624][65536] packed k-pairs
    const float* __restrict__ bias1,
    uint32_t* __restrict__ hP) {              // [256][65536] packed k-pairs
  __shared__ unsigned short lA[2][128 * 40];
  __shared__ uint32_t       lB[2][128 * BCSTRIDE];

  const int tid = threadIdx.x;
  const int lane = tid & 31;
  const int laneLo = lane & 15;
  const int laneHi = (lane >> 4) & 1;
  const int wid = tid >> 5;
  const int wave_m = wid & 3;                 // * 32 rows
  const int wave_n = wid >> 2;                // * 64 cols
  const int n0 = blockIdx.x * 128;
  const int m0 = blockIdx.y * 128;

  v8f acc[2][4] = {};

  auto stage = [&](int k0, int buf) {
    // A: 128 rows x 32 bf16, 512 16-B chunks
    #pragma unroll
    for (int i = 0; i < 2; ++i) {
      int idx = tid + i * 256;
      int row = idx >> 2, seg = idx & 3;
      const unsigned short* g = W1b + (size_t)(m0 + row) * KDIM + k0 + seg * 8;
      unsigned short* l = &lA[buf][row * 40 + seg * 8];
#if USE_ASYNC
      __builtin_amdgcn_global_load_async_to_lds_b128(AS1_B128(g), AS3_B128(l), 0, 0);
#else
      *(uint4*)l = *(const uint4*)g;
#endif
    }
    // B: gather one column's 16 kpair rows per thread-pair -> [col][kpair]
    {
      int q0 = k0 >> 1;
      int c = tid >> 1, h = tid & 1;
      const uint32_t* g = featP + (size_t)(q0 + h * 8) * NPIX_TOT + n0 + c;
      uint32_t* l = &lB[buf][c * BCSTRIDE + h * 8];
#if USE_ASYNC
      #pragma unroll
      for (int j = 0; j < 8; ++j)
        __builtin_amdgcn_global_load_async_to_lds_b32(
            AS1_B32(g + (size_t)j * NPIX_TOT), AS3_B32(l + j), 0, 0);
#else
      uint32_t w[8];
      #pragma unroll
      for (int j = 0; j < 8; ++j) w[j] = g[(size_t)j * NPIX_TOT];
      *(uint4*)(l)     = *(uint4*)&w[0];
      *(uint4*)(l + 4) = *(uint4*)&w[4];
#endif
    }
  };

  stage(0, 0);
  async_wait0();
  __syncthreads();

  int buf = 0;
  for (int k0 = 0; k0 < KDIM; k0 += 32) {
    if (k0 + 32 < KDIM) stage(k0 + 32, buf ^ 1);

    v16bf afrag[2];
    #pragma unroll
    for (int i = 0; i < 2; ++i) {
      const unsigned short* base = &lA[buf][(wave_m * 32 + i * 16 + laneLo) * 40];
      union { uint4 q[2]; v16bf v; } u;
      u.q[0] = *(const uint4*)(base + laneHi * 16);     // K 0..7  (hi lanes 16..23)
      u.q[1] = *(const uint4*)(base + laneHi * 16 + 8); // K 8..15 (hi 24..31)
      afrag[i] = u.v;
    }
    v16bf bfrag[4];
    #pragma unroll
    for (int j = 0; j < 4; ++j) {
      const uint32_t* base =
          &lB[buf][(wave_n * 64 + j * 16 + laneLo) * BCSTRIDE + laneHi * 8];
      union { uint4 q[2]; v16bf v; } u;
      u.q[0] = *(const uint4*)(base);
      u.q[1] = *(const uint4*)(base + 4);
      bfrag[j] = u.v;
    }
    #pragma unroll
    for (int i = 0; i < 2; ++i)
      #pragma unroll
      for (int j = 0; j < 4; ++j)
        acc[i][j] = __builtin_amdgcn_wmma_f32_16x16x32_bf16(
            false, afrag[i], false, bfrag[j], (short)0, acc[i][j], false, false);

    async_wait0();
    __syncthreads();
    buf ^= 1;
  }

  // epilogue: +bias, ReLU, pack adjacent D rows (r, r+1) into bf16 pairs
  #pragma unroll
  for (int i = 0; i < 2; ++i) {
    int mbase = m0 + wave_m * 32 + i * 16 + laneHi * 8;
    #pragma unroll
    for (int j = 0; j < 4; ++j) {
      int n = n0 + wave_n * 64 + j * 16 + laneLo;
      #pragma unroll
      for (int r = 0; r < 8; r += 2) {
        float v0 = fmaxf(acc[i][j][r]     + bias1[mbase + r],     0.f);
        float v1 = fmaxf(acc[i][j][r + 1] + bias1[mbase + r + 1], 0.f);
        hP[(size_t)((mbase + r) >> 1) * NPIX_TOT + n] = packbf2(v0, v1);
      }
    }
  }
}

// --------------------------- GEMM2: 96 x 65536 x 512 -----------------------
// Block tile 96(M) x 128(N); 8 waves each own 16 cols and full M (6 WMMA).
__global__ __launch_bounds__(256) void gemm2_kernel(
    const unsigned short* __restrict__ W2p,   // [96][512] bf16 (BN2-folded)
    const uint32_t* __restrict__ hP,          // [256][65536]
    const float* __restrict__ b2p,
    float* __restrict__ out) {                // [4][96][128][128]
  __shared__ unsigned short lA[2][96 * 40];
  __shared__ uint32_t       lB[2][128 * BCSTRIDE];

  const int tid = threadIdx.x;
  const int lane = tid & 31;
  const int laneLo = lane & 15;
  const int laneHi = (lane >> 4) & 1;
  const int wid = tid >> 5;                   // col group * 16
  const int n0 = blockIdx.x * 128;

  v8f acc[6] = {};

  auto stage = [&](int k0, int buf) {
    #pragma unroll
    for (int i = 0; i < 2; ++i) {
      int idx = tid + i * 256;                // need 384 16-B chunks
      if (idx < 384) {
        int row = idx >> 2, seg = idx & 3;
        const unsigned short* g = W2p + (size_t)row * HDIM + k0 + seg * 8;
        unsigned short* l = &lA[buf][row * 40 + seg * 8];
#if USE_ASYNC
        __builtin_amdgcn_global_load_async_to_lds_b128(AS1_B128(g), AS3_B128(l), 0, 0);
#else
        *(uint4*)l = *(const uint4*)g;
#endif
      }
    }
    {
      int q0 = k0 >> 1;
      int c = tid >> 1, h = tid & 1;
      const uint32_t* g = hP + (size_t)(q0 + h * 8) * NPIX_TOT + n0 + c;
      uint32_t* l = &lB[buf][c * BCSTRIDE + h * 8];
#if USE_ASYNC
      #pragma unroll
      for (int j = 0; j < 8; ++j)
        __builtin_amdgcn_global_load_async_to_lds_b32(
            AS1_B32(g + (size_t)j * NPIX_TOT), AS3_B32(l + j), 0, 0);
#else
      uint32_t w[8];
      #pragma unroll
      for (int j = 0; j < 8; ++j) w[j] = g[(size_t)j * NPIX_TOT];
      *(uint4*)(l)     = *(uint4*)&w[0];
      *(uint4*)(l + 4) = *(uint4*)&w[4];
#endif
    }
  };

  stage(0, 0);
  async_wait0();
  __syncthreads();

  int buf = 0;
  for (int k0 = 0; k0 < HDIM; k0 += 32) {
    if (k0 + 32 < HDIM) stage(k0 + 32, buf ^ 1);

    v16bf bfrag;
    {
      const uint32_t* base = &lB[buf][(wid * 16 + laneLo) * BCSTRIDE + laneHi * 8];
      union { uint4 q[2]; v16bf v; } u;
      u.q[0] = *(const uint4*)(base);
      u.q[1] = *(const uint4*)(base + 4);
      bfrag = u.v;
    }
    #pragma unroll
    for (int i = 0; i < 6; ++i) {
      const unsigned short* base = &lA[buf][(i * 16 + laneLo) * 40];
      union { uint4 q[2]; v16bf v; } u;
      u.q[0] = *(const uint4*)(base + laneHi * 16);
      u.q[1] = *(const uint4*)(base + laneHi * 16 + 8);
      acc[i] = __builtin_amdgcn_wmma_f32_16x16x32_bf16(
          false, u.v, false, bfrag, (short)0, acc[i], false, false);
    }

    async_wait0();
    __syncthreads();
    buf ^= 1;
  }

  int n = n0 + wid * 16 + laneLo;
  int b = n >> 14, pix = n & (IMG - 1);
  #pragma unroll
  for (int i = 0; i < 6; ++i) {
    int mbase = i * 16 + laneHi * 8;
    #pragma unroll
    for (int r = 0; r < 8; ++r) {
      int m = mbase + r;
      out[((size_t)b * ODIM + m) * IMG + pix] = acc[i][r] + b2p[m];
    }
  }
}

// ---------------------------------------------------------------------------
extern "C" void kernel_launch(void* const* d_in, const int* in_sizes, int n_in,
                              void* d_out, int out_size, void* d_ws, size_t ws_size,
                              hipStream_t stream) {
  const float* x      = (const float*)d_in[0];
  const float* bn1_g  = (const float*)d_in[1];
  const float* bn1_b  = (const float*)d_in[2];
  const float* bn1_m  = (const float*)d_in[3];
  const float* bn1_v  = (const float*)d_in[4];
  const float* w1     = (const float*)d_in[5];
  const float* cb1    = (const float*)d_in[6];
  const float* bn2_g  = (const float*)d_in[7];
  const float* bn2_b  = (const float*)d_in[8];
  const float* bn2_m  = (const float*)d_in[9];
  const float* bn2_v  = (const float*)d_in[10];
  const float* w2     = (const float*)d_in[11];
  const float* cb2    = (const float*)d_in[12];
  float* out = (float*)d_out;

  char* ws = (char*)d_ws;
  const size_t FEAT_BYTES = (size_t)QDIM * NPIX_TOT * 4;   // 163,577,856
  const size_t HP_BYTES   = (size_t)HQ   * NPIX_TOT * 4;   // 67,108,864
  const size_t W1B_BYTES  = (size_t)HDIM * KDIM * 2;       // 1,277,952
  const size_t W2P_BYTES  = (size_t)ODIM * HDIM * 2;       // 98,304
  uint32_t*       featP = (uint32_t*)ws;
  uint32_t*       hP    = (uint32_t*)(ws + FEAT_BYTES);
  unsigned short* W1b   = (unsigned short*)(ws + FEAT_BYTES + HP_BYTES);
  unsigned short* W2p   = (unsigned short*)(ws + FEAT_BYTES + HP_BYTES + W1B_BYTES);
  float*          b2p   = (float*)(ws + FEAT_BYTES + HP_BYTES + W1B_BYTES + W2P_BYTES);

  prep_w1<<<(HDIM * KDIM + 255) / 256, 256, 0, stream>>>(w1, W1b);
  prep_w2<<<(ODIM * HDIM + 255) / 256, 256, 0, stream>>>(w2, bn2_g, bn2_v, W2p);
  prep_b2<<<1, 128, 0, stream>>>(w2, bn2_g, bn2_b, bn2_m, bn2_v, cb2, b2p);

  features_kernel<<<dim3(48, 4), 256, 2 * IMGSZ * sizeof(float), stream>>>(
      x, bn1_g, bn1_b, bn1_m, bn1_v, featP);

  gemm1_kernel<<<dim3(NPIX_TOT / 128, 4), 256, 0, stream>>>(W1b, featP, cb1, hP);
  gemm2_kernel<<<dim3(NPIX_TOT / 128), 256, 0, stream>>>(W2p, hP, b2p, out);
}
